// Attention_30485677867708
// MI455X (gfx1250) — matrix operand order repaired and checked by reading, not compiled, not run on hardware
//
#include <hip/hip_runtime.h>

#define NROWS 8192
#define DIN   256
#define HDIM  128
#define NSPLIT 4
#define KSPAN (NROWS / NSPLIT)   // 2048 keys per split

typedef __bf16 v16bf __attribute__((ext_vector_type(16)));
typedef float  v8f   __attribute__((ext_vector_type(8)));

__device__ __forceinline__ v8f wmma_bf16(v16bf a, v16bf b, v8f c) {
  // D = A(16x32 bf16) * B(32x16 bf16) + C(16x16 f32)
  return __builtin_amdgcn_wmma_f32_16x16x32_bf16(false, a, false, b, (short)0, c, false, false);
}

union FragU { v16bf v; uint4 u[2]; };

// A fragment (16x32, M x K): row = lane%16, halves i<8 -> K=k0+hi*8+i, i>=8 -> K=k0+16+hi*8+(i-8)
__device__ __forceinline__ v16bf load_frag_A(const __bf16* __restrict__ base, int ld, int k0) {
  const int lane = threadIdx.x & 31;
  const int m = lane & 15, hi = lane >> 4;
  const __bf16* row = base + (size_t)m * ld;
  FragU r;
  r.u[0] = *(const uint4*)(row + k0 + hi * 8);
  r.u[1] = *(const uint4*)(row + k0 + 16 + hi * 8);
  return r.v;
}

// B fragment (32x16, K x N) from row-major S[n][k] (B = S^T): col = lane%16, i<8 -> K=k0+hi*8+i, i>=8 -> K=k0+16+hi*8+(i-8)
__device__ __forceinline__ v16bf load_frag_B(const __bf16* __restrict__ base, int ld, int k0) {
  const int lane = threadIdx.x & 31;
  const int n = lane & 15, hi = lane >> 4;
  const __bf16* row = base + (size_t)n * ld;
  FragU r;
  r.u[0] = *(const uint4*)(row + k0 + hi * 8);
  r.u[1] = *(const uint4*)(row + k0 + 16 + hi * 8);
  return r.v;
}

// load all 8 B fragments (2 key-tiles x 4 K-frags) for a 32-key block
__device__ __forceinline__ void load_kblock(const __bf16* __restrict__ kptr, int kb,
                                            v16bf* __restrict__ B) {
#pragma unroll
  for (int f = 0; f < 4; ++f) {
    B[f]     = load_frag_B(kptr + (size_t)kb * HDIM, HDIM, f * 32);
    B[4 + f] = load_frag_B(kptr + (size_t)(kb + 16) * HDIM, HDIM, f * 32);
  }
}

// ---------------- prep: f32 -> bf16 conversions and weight transposes ----------------
__global__ void prep_kernel(const float* __restrict__ input, const float* __restrict__ Q,
                            const float* __restrict__ K, const float* __restrict__ Vd,
                            const float* __restrict__ Vu,
                            __bf16* __restrict__ inB, __bf16* __restrict__ QT,
                            __bf16* __restrict__ KT, __bf16* __restrict__ VDT,
                            __bf16* __restrict__ VUT) {
  const int stride = gridDim.x * blockDim.x;
  const int tid = blockIdx.x * blockDim.x + threadIdx.x;
  for (int i = tid; i < NROWS * DIN; i += stride) inB[i] = (__bf16)input[i];
  for (int i = tid; i < DIN * HDIM; i += stride) {
    const int in = i / HDIM, on = i % HDIM;       // Q[in][on] -> QT[on][in]
    QT[on * DIN + in]  = (__bf16)Q[i];
    KT[on * DIN + in]  = (__bf16)K[i];
    VDT[on * DIN + in] = (__bf16)Vd[i];
  }
  for (int i = tid; i < HDIM * DIN; i += stride) {
    const int h = i / DIN, n = i % DIN;           // Vu[h][n] -> VUT[n][h]
    VUT[n * HDIM + h] = (__bf16)Vu[i];
  }
}

// ---------------- projections: one (row-tile, weight) pair per wave; 1536 waves ----------------
__global__ __launch_bounds__(128) void proj_kernel(const __bf16* __restrict__ inB,
                                                   const __bf16* __restrict__ QT,
                                                   const __bf16* __restrict__ KT,
                                                   const __bf16* __restrict__ VDT,
                                                   __bf16* __restrict__ q, __bf16* __restrict__ k,
                                                   __bf16* __restrict__ vT) {
  const int w = blockIdx.x * 4 + (threadIdx.x >> 5);
  const int blk = w & 511;                        // 512 row-tiles of 16
  const int wt = w >> 9;                          // 0=Q, 1=K, 2=Vd
  const int row0 = blk * 16;
  const int lane = threadIdx.x & 31;
  const int m = lane & 15, hi = lane >> 4;
  const __bf16* W = (wt == 0) ? QT : ((wt == 1) ? KT : VDT);

  v16bf A[8];
#pragma unroll
  for (int f = 0; f < 8; ++f) A[f] = load_frag_A(inB + (size_t)row0 * DIN, DIN, f * 32);

#pragma unroll
  for (int c = 0; c < 8; ++c) {
    v16bf B[8];
#pragma unroll
    for (int f = 0; f < 8; ++f) B[f] = load_frag_B(W + (size_t)(c * 16) * DIN, DIN, f * 32);
    v8f acc = {};
#pragma unroll
    for (int f = 0; f < 8; ++f) acc = wmma_bf16(A[f], B[f], acc);
#pragma unroll
    for (int r = 0; r < 8; ++r) {
      const int rr = row0 + r + 8 * hi;           // C-layout: VGPR r holds row r + 8*hi, col = lane%16
      const int cc = c * 16 + m;
      if (wt == 0)      q[(size_t)rr * HDIM + cc] = (__bf16)acc[r];
      else if (wt == 1) k[(size_t)rr * HDIM + cc] = (__bf16)acc[r];
      else              vT[(size_t)cc * NROWS + rr] = (__bf16)acc[r];
    }
  }
}

// ---------------- split-K flash attention: partial O + (m,l) per split ----------------
__global__ __launch_bounds__(128) void flash_kernel(const __bf16* __restrict__ q,
                                                    const __bf16* __restrict__ k,
                                                    const __bf16* __restrict__ vT,
                                                    float* __restrict__ Opart,
                                                    float* __restrict__ Mpart,
                                                    float* __restrict__ Lpart) {
  __shared__ __align__(16) __bf16 Psh[4][16 * 32];
  const int wave = threadIdx.x >> 5;
  const int w = blockIdx.x * 4 + wave;            // 2048 waves
  const int blk = w >> 2;                         // row-tile 0..511
  const int split = w & 3;                        // key split 0..3
  const int row0 = blk * 16;
  const int kbeg = split * KSPAN, kend = kbeg + KSPAN;
  const int lane = threadIdx.x & 31;
  const int m = lane & 15, hi = lane >> 4;
  __bf16* P = &Psh[wave][0];

  v16bf Aq[4];
#pragma unroll
  for (int f = 0; f < 4; ++f) Aq[f] = load_frag_A(q + (size_t)row0 * HDIM, HDIM, f * 32);

  v8f O[8];
#pragma unroll
  for (int c = 0; c < 8; ++c) O[c] = (v8f){};
  float mrow[8], lrow[8];
#pragma unroll
  for (int r = 0; r < 8; ++r) { mrow[r] = -__builtin_inff(); lrow[r] = 0.0f; }

  auto flash_step = [&](int kb, v16bf* __restrict__ Bk, v16bf* __restrict__ Bnext, int kbn) {
    // S = Q @ K^T : 8 back-to-back WMMAs on pre-loaded fragments
    v8f s0 = {}, s1 = {};
#pragma unroll
    for (int f = 0; f < 4; ++f) {
      s0 = wmma_bf16(Aq[f], Bk[f], s0);
      s1 = wmma_bf16(Aq[f], Bk[4 + f], s1);
    }
    // issue V-block loads early so softmax VALU work hides them
    v16bf Bv[8];
#pragma unroll
    for (int c = 0; c < 8; ++c)
      Bv[c] = load_frag_B(vT + (size_t)(c * 16) * NROWS, NROWS, kb);
    // prefetch next K block (consumed next step)
    load_kblock(k, kbn, Bnext);

    float alpha[8];
#pragma unroll
    for (int r = 0; r < 8; ++r) {
      // row r + 8*hi is striped over the 16 lanes of this half-wave
      float mx = fmaxf(s0[r], s1[r]);
#pragma unroll
      for (int off = 8; off > 0; off >>= 1) mx = fmaxf(mx, __shfl_xor(mx, off, 32));
      const float mnew = fmaxf(mrow[r], mx);
      alpha[r] = __expf(mrow[r] - mnew);
      mrow[r] = mnew;
      const float p0 = __expf(s0[r] - mnew);
      const float p1 = __expf(s1[r] - mnew);
      float ls = p0 + p1;
#pragma unroll
      for (int off = 8; off > 0; off >>= 1) ls += __shfl_xor(ls, off, 32);
      lrow[r] = lrow[r] * alpha[r] + ls;
      const int mm = r + 8 * hi;                  // stage P (16x32 bf16) for C->A re-layout
      P[mm * 32 + m] = (__bf16)p0;
      P[mm * 32 + 16 + m] = (__bf16)p1;
    }
#pragma unroll
    for (int c = 0; c < 8; ++c)
#pragma unroll
      for (int r = 0; r < 8; ++r) O[c][r] *= alpha[r];

    const __bf16* prow = P + m * 32;              // intra-wave LDS roundtrip (DS in-order)
    FragU ap;
    ap.u[0] = *(const uint4*)(prow + hi * 8);
    ap.u[1] = *(const uint4*)(prow + 16 + hi * 8);
#pragma unroll
    for (int c = 0; c < 8; ++c) O[c] = wmma_bf16(ap.v, Bv[c], O[c]);
  };

  v16bf Bk0[8], Bk1[8];
  load_kblock(k, kbeg, Bk0);
  for (int kb = kbeg; kb < kend; kb += 64) {
    flash_step(kb, Bk0, Bk1, kb + 32);
    flash_step(kb + 32, Bk1, Bk0, (kb + 64 < kend) ? kb + 64 : kbeg);
  }

  // store unnormalized partial O (f32) + per-row stats
  float* Od = Opart + (size_t)split * NROWS * HDIM;
#pragma unroll
  for (int c = 0; c < 8; ++c)
#pragma unroll
    for (int r = 0; r < 8; ++r)
      Od[(size_t)(row0 + r + 8 * hi) * HDIM + c * 16 + m] = O[c][r];
  if (m == 0) {
#pragma unroll
    for (int r = 0; r < 8; ++r) {
      const int rr = row0 + r + 8 * hi;
      Mpart[split * NROWS + rr] = mrow[r];
      Lpart[split * NROWS + rr] = lrow[r];
    }
  }
}

// ---------------- merge split-K partials: log-sum-exp combine -> bf16 O ----------------
__global__ void merge_kernel(const float* __restrict__ Opart, const float* __restrict__ Mpart,
                             const float* __restrict__ Lpart, __bf16* __restrict__ o) {
  const int idx = blockIdx.x * blockDim.x + threadIdx.x;   // over N*H
  if (idx >= NROWS * HDIM) return;
  const int row = idx / HDIM;
  float M = -__builtin_inff();
#pragma unroll
  for (int s = 0; s < NSPLIT; ++s) M = fmaxf(M, Mpart[s * NROWS + row]);
  float L = 0.0f, acc = 0.0f;
#pragma unroll
  for (int s = 0; s < NSPLIT; ++s) {
    const float sc = __expf(Mpart[s * NROWS + row] - M);
    L += Lpart[s * NROWS + row] * sc;
    acc += Opart[(size_t)s * NROWS * HDIM + idx] * sc;
  }
  o[idx] = (__bf16)(acc / L);
}

// ---------------- output projection: dE = O @ V_up ; (row-tile, col-half) per wave ----------------
__global__ __launch_bounds__(128) void outproj_kernel(const __bf16* __restrict__ o,
                                                      const __bf16* __restrict__ VUT,
                                                      float* __restrict__ out) {
  const int w = blockIdx.x * 4 + (threadIdx.x >> 5); // 1024 waves
  const int blk = w >> 1;
  const int chalf = (w & 1) * 8;
  const int row0 = blk * 16;
  const int lane = threadIdx.x & 31;
  const int m = lane & 15, hi = lane >> 4;

  v16bf A[4];
#pragma unroll
  for (int f = 0; f < 4; ++f) A[f] = load_frag_A(o + (size_t)row0 * HDIM, HDIM, f * 32);

#pragma unroll
  for (int ci = 0; ci < 8; ++ci) {
    const int c = chalf + ci;
    v16bf B[4];
#pragma unroll
    for (int f = 0; f < 4; ++f) B[f] = load_frag_B(VUT + (size_t)(c * 16) * HDIM, HDIM, f * 32);
    v8f acc = {};
#pragma unroll
    for (int f = 0; f < 4; ++f) acc = wmma_bf16(A[f], B[f], acc);
#pragma unroll
    for (int r = 0; r < 8; ++r)
      out[(size_t)(row0 + r + 8 * hi) * DIN + c * 16 + m] = acc[r];
  }
}

extern "C" void kernel_launch(void* const* d_in, const int* in_sizes, int n_in,
                              void* d_out, int out_size, void* d_ws, size_t ws_size,
                              hipStream_t stream) {
  const float* input = (const float*)d_in[0];
  const float* Q     = (const float*)d_in[1];
  const float* K     = (const float*)d_in[2];
  const float* Vd    = (const float*)d_in[3];
  const float* Vu    = (const float*)d_in[4];
  float* out = (float*)d_out;

  __bf16* ws  = (__bf16*)d_ws;
  __bf16* inB = ws;                       // N*D
  __bf16* QT  = inB + (size_t)NROWS * DIN;
  __bf16* KT  = QT + (size_t)HDIM * DIN;
  __bf16* VDT = KT + (size_t)HDIM * DIN;
  __bf16* VUT = VDT + (size_t)HDIM * DIN; // D*H
  __bf16* qb  = VUT + (size_t)DIN * HDIM;
  __bf16* kb  = qb + (size_t)NROWS * HDIM;
  __bf16* vT  = kb + (size_t)NROWS * HDIM;
  __bf16* ob  = vT + (size_t)HDIM * NROWS;
  __bf16* bf16_end = ob + (size_t)NROWS * HDIM;

  float* Opart = (float*)bf16_end;        // NSPLIT*N*H f32 (16 MB)
  float* Mpart = Opart + (size_t)NSPLIT * NROWS * HDIM;
  float* Lpart = Mpart + (size_t)NSPLIT * NROWS;

  prep_kernel<<<2048, 256, 0, stream>>>(input, Q, K, Vd, Vu, inB, QT, KT, VDT, VUT);
  proj_kernel<<<384, 128, 0, stream>>>(inB, QT, KT, VDT, qb, kb, vT);
  flash_kernel<<<512, 128, 0, stream>>>(qb, kb, vT, Opart, Mpart, Lpart);
  merge_kernel<<<(NROWS * HDIM + 255) / 256, 256, 0, stream>>>(Opart, Mpart, Lpart, ob);
  outproj_kernel<<<256, 128, 0, stream>>>(ob, VUT, out);
}
